// LorentzAttention_31791347925117
// MI455X (gfx1250) — matrix operand order repaired and checked by reading, not compiled, not run on hardware
//
#include <hip/hip_runtime.h>

typedef __bf16 bf16;
typedef __attribute__((ext_vector_type(16))) __bf16 v16bf;
typedef __attribute__((ext_vector_type(8)))  __bf16 v8bf;
typedef __attribute__((ext_vector_type(8)))  float  v8f;
typedef int v4i_g __attribute__((vector_size(16)));   // pointee type for async-LDS builtin

#define B_    4
#define N_    2048
#define D_    513
#define H_    8
#define DH    64
#define DP    544    // padded input dim (17*32)
#define DQK   96     // padded q~/k~ dim (3*32)
#define DV    80     // padded v dim
#define VSTR  72     // LDS V-buffer key stride (16B aligned rows, bank-spread)
#define INVK  10.0f
#define KCURV 0.1f
#define EPSF  1e-9f
#define SCALE 0.125f
#define SM2   (0.125f * 1.4426950408889634f)  // SCALE * log2(e)

#if __has_builtin(__builtin_amdgcn_global_load_async_to_lds_b128) && \
    __has_builtin(__builtin_amdgcn_s_wait_asynccnt)
#define HAS_ASYNC 1
#else
#define HAS_ASYNC 0
#endif

// ---------- helpers ----------
__device__ __forceinline__ v8bf ld16(const bf16* p) {
  return *reinterpret_cast<const v8bf*>(p);
}
__device__ __forceinline__ v16bf ld32(const bf16* p) {
  v8bf lo = *reinterpret_cast<const v8bf*>(p);
  v8bf hi = *reinterpret_cast<const v8bf*>(p + 8);
  return __builtin_shufflevector(lo, hi, 0,1,2,3,4,5,6,7,8,9,10,11,12,13,14,15);
}
__device__ __forceinline__ v16bf ldA2(const bf16* p0, const bf16* p1) {
  v8bf lo = *reinterpret_cast<const v8bf*>(p0);
  v8bf hi = *reinterpret_cast<const v8bf*>(p1);
  return __builtin_shufflevector(lo, hi, 0,1,2,3,4,5,6,7,8,9,10,11,12,13,14,15);
}
__device__ __forceinline__ v8f wmma_bf16(v16bf a, v16bf b, v8f c) {
  return __builtin_amdgcn_wmma_f32_16x16x32_bf16(false, a, false, b, (short)0, c, false, false);
}
__device__ __forceinline__ float redmax16(float v) {
#pragma unroll
  for (int m = 8; m >= 1; m >>= 1) v = fmaxf(v, __shfl_xor(v, m, 16));
  return v;
}
__device__ __forceinline__ float redsum16(float v) {
#pragma unroll
  for (int m = 8; m >= 1; m >>= 1) v += __shfl_xor(v, m, 16);
  return v;
}
// async global->LDS 16B copy (per-lane addresses), ASYNCcnt-tracked
__device__ __forceinline__ void cp16_g2l(bf16* l, const bf16* g) {
#if HAS_ASYNC
  __builtin_amdgcn_global_load_async_to_lds_b128(
      (__attribute__((address_space(1))) v4i_g*)g,
      (__attribute__((address_space(3))) v4i_g*)l, 0, 0);
#else
  *reinterpret_cast<v8bf*>(l) = ld16(g);
#endif
}
__device__ __forceinline__ void await_async() {
#if HAS_ASYNC
  __builtin_amdgcn_s_wait_asynccnt(0);
#endif
}

// ---------- packing kernels ----------
__global__ void k_pack_x(const float* __restrict__ x, bf16* __restrict__ xb, int total) {
  int i = blockIdx.x * 256 + threadIdx.x;
  if (i >= total) return;
  int d = i % DP;
  int n = i / DP;  // flattened (b,n)
  xb[i] = (d < D_) ? (bf16)x[(size_t)n * D_ + d] : (bf16)0.0f;
}

// w: [H][513][64] f32 -> wt: [H][64][544] bf16 (transposed + zero padded)
__global__ void k_pack_w(const float* __restrict__ w, bf16* __restrict__ wt, int total) {
  int i = blockIdx.x * 256 + threadIdx.x;
  if (i >= total) return;
  int d = i % DP;
  int r = i / DP;
  int m = r % DH;
  int h = r / DH;
  wt[i] = (d < D_) ? (bf16)w[((size_t)h * D_ + d) * DH + m] : (bf16)0.0f;
}

// wo: [513][512] f32 -> wot: [512][544] bf16.  K remap: k<512 -> Wo[1+k][m],
// k==543 -> Wo[0][m] (time row), 512..542 -> 0.
__global__ void k_pack_wo(const float* __restrict__ wo, bf16* __restrict__ wot, int total) {
  int i = blockIdx.x * 256 + threadIdx.x;
  if (i >= total) return;
  int d = i % DP;
  int m = i / DP;
  float v = 0.0f;
  if (d < 512)          v = wo[(size_t)(1 + d) * 512 + m];
  else if (d == DP - 1) v = wo[m];
  wot[i] = (bf16)v;
}

// ---------- kernel 1: QKV projections + Lorentz lift ----------
// grid (N/64, H, B), block 128 (4 waves).  Each wave: 16 rows x 64 cols x K=544.
__global__ __launch_bounds__(128) void k_proj(
    const bf16* __restrict__ xb,
    const bf16* __restrict__ wqT, const bf16* __restrict__ wkT, const bf16* __restrict__ wvT,
    const float* __restrict__ bq, const float* __restrict__ bk, const float* __restrict__ bv,
    bf16* __restrict__ qt, bf16* __restrict__ ktb, bf16* __restrict__ vtb) {
  __shared__ alignas(16) bf16 Xblk[64 * 288];
  int b = blockIdx.z, h = blockIdx.y;
  int n0 = blockIdx.x * 64;
  int tid = threadIdx.x;
  int wid = tid >> 5, lane = tid & 31, col = lane & 15, hi = lane >> 4;

  v8f accQ[4] = {}, accK[4] = {}, accV[4] = {};

  const int kbase_[2] = {0, 256};
  const int klen_[2]  = {256, 288};
#pragma unroll
  for (int s = 0; s < 2; ++s) {
    int kbase = kbase_[s], klen = klen_[s];
    __syncthreads();
    int nchunk = klen / 8;
    for (int idx = tid; idx < 64 * nchunk; idx += 128) {
      int row = idx / nchunk, c = idx % nchunk;
      cp16_g2l(&Xblk[row * klen + c * 8],
               &xb[((size_t)b * N_ + n0 + row) * DP + kbase + c * 8]);
    }
    await_async();
    __syncthreads();
    int lrow = wid * 16 + col;
    for (int kk = 0; kk < klen / 32; ++kk) {
      int kb = kk * 32;
      v16bf a = ldA2(&Xblk[lrow * klen + kb + hi * 16],
                     &Xblk[lrow * klen + kb + hi * 16 + 8]);
#pragma unroll
      for (int t = 0; t < 4; ++t) {
        int ncol = t * 16 + col;
        size_t wrow = ((size_t)h * DH + ncol) * DP + kbase + kb + hi * 16;
        accQ[t] = wmma_bf16(a, ld32(&wqT[wrow]), accQ[t]);
        accK[t] = wmma_bf16(a, ld32(&wkT[wrow]), accK[t]);
        accV[t] = wmma_bf16(a, ld32(&wvT[wrow]), accV[t]);
      }
    }
  }

  size_t bh = (size_t)(b * H_ + h);
  // ---- Q epilogue: q~ = [space(64), +t, 0 x31] ----
  {
    float val[4][8], s2[8];
#pragma unroll
    for (int r = 0; r < 8; ++r) s2[r] = 0.0f;
#pragma unroll
    for (int t = 0; t < 4; ++t) {
      float bb = bq[h * DH + t * 16 + col];
#pragma unroll
      for (int r = 0; r < 8; ++r) { float v = accQ[t][r] + bb; val[t][r] = v; s2[r] += v * v; }
    }
#pragma unroll
    for (int r = 0; r < 8; ++r) {
      float tq = sqrtf(fmaxf(INVK + redsum16(s2[r]), EPSF));
      int row = n0 + wid * 16 + r + 8 * hi;
      bf16* qr = qt + (bh * N_ + row) * DQK;
      for (int t = 0; t < 4; ++t) qr[t * 16 + col] = (bf16)val[t][r];
      qr[64 + col] = (col == 0) ? (bf16)tq : (bf16)0.0f;
      qr[80 + col] = (bf16)0.0f;
    }
  }
  // ---- K epilogue: k~ = [space(64), -t, 0 x31] ----
  {
    float val[4][8], s2[8];
#pragma unroll
    for (int r = 0; r < 8; ++r) s2[r] = 0.0f;
#pragma unroll
    for (int t = 0; t < 4; ++t) {
      float bb = bk[h * DH + t * 16 + col];
#pragma unroll
      for (int r = 0; r < 8; ++r) { float v = accK[t][r] + bb; val[t][r] = v; s2[r] += v * v; }
    }
#pragma unroll
    for (int r = 0; r < 8; ++r) {
      float tk = sqrtf(fmaxf(INVK + redsum16(s2[r]), EPSF));
      int row = n0 + wid * 16 + r + 8 * hi;
      bf16* kr = ktb + (bh * N_ + row) * DQK;
      for (int t = 0; t < 4; ++t) kr[t * 16 + col] = (bf16)val[t][r];
      kr[64 + col] = (col == 0) ? (bf16)(-tk) : (bf16)0.0f;
      kr[80 + col] = (bf16)0.0f;
    }
  }
  // ---- V epilogue: v = [t, space(64), 0 x15] ----
  {
    float val[4][8], s2[8];
#pragma unroll
    for (int r = 0; r < 8; ++r) s2[r] = 0.0f;
#pragma unroll
    for (int t = 0; t < 4; ++t) {
      float bb = bv[h * DH + t * 16 + col];
#pragma unroll
      for (int r = 0; r < 8; ++r) { float v = accV[t][r] + bb; val[t][r] = v; s2[r] += v * v; }
    }
#pragma unroll
    for (int r = 0; r < 8; ++r) {
      float tv = sqrtf(fmaxf(INVK + redsum16(s2[r]), EPSF));
      int row = n0 + wid * 16 + r + 8 * hi;
      bf16* vr = vtb + (bh * N_ + row) * DV;
      for (int t = 0; t < 4; ++t) vr[1 + t * 16 + col] = (bf16)val[t][r];
      if (col == 0) vr[0] = (bf16)tv;
      else          vr[64 + col] = (bf16)0.0f;  // dims 65..79
    }
  }
}

// ---------- kernel 2: flash Lorentz attention (double-buffered, async K staging) ----------
// grid (N/128, H, B), block 256 (8 waves).  wave = 16-query tile, streams 64-key blocks.
__global__ __launch_bounds__(256) void k_attn(
    const bf16* __restrict__ qt, const bf16* __restrict__ ktb, const bf16* __restrict__ vtb,
    bf16* __restrict__ catp, float* __restrict__ tsum) {
  __shared__ alignas(16) bf16 Kbuf[2][64 * DQK];   // [key][96]
  __shared__ alignas(16) bf16 Vbuf[2][DV * VSTR];  // [dim][key] transposed
  __shared__ alignas(16) bf16 Pscr[8 * 16 * 32];   // per-wave P scratch
  int b = blockIdx.z, h = blockIdx.y;
  int tid = threadIdx.x;
  int wid = tid >> 5, lane = tid & 31, col = lane & 15, hi = lane >> 4;
  size_t bh = (size_t)(b * H_ + h);
  int q0 = (blockIdx.x * 8 + wid) * 16;

  v16bf qf[3];
  {
    const bf16* qr = qt + (bh * N_ + q0 + col) * DQK;
#pragma unroll
    for (int f = 0; f < 3; ++f)
      qf[f] = ldA2(qr + f * 32 + hi * 16, qr + f * 32 + hi * 16 + 8);
  }
  v8f o[5] = {};
  float mrow[8], lrow[8];
#pragma unroll
  for (int r = 0; r < 8; ++r) { mrow[r] = -1e30f; lrow[r] = 0.0f; }

  v8bf vreg[3];
  // issue async K block copy: 64x96 bf16 = 768 x 16B chunks = 3/thread
  auto issueK = [&](int kb, int buf) {
#pragma unroll
    for (int j = 0; j < 3; ++j) {
      int idx = tid + j * 256;
      int key = idx / 12, c = idx % 12;
      cp16_g2l(&Kbuf[buf][key * DQK + c * 8],
               &ktb[(bh * N_ + kb * 64 + key) * DQK + c * 8]);
    }
  };
  // prefetch V block into registers: 64x80 bf16 = 640 x 16B chunks
  auto loadV = [&](int kb) {
#pragma unroll
    for (int j = 0; j < 3; ++j) {
      int idx = tid + j * 256;
      if (idx < 640) {
        int key = idx / 10, c = idx % 10;
        vreg[j] = ld16(&vtb[(bh * N_ + kb * 64 + key) * DV + c * 8]);
      }
    }
  };
  // scatter V registers into LDS transposed [dim][key]
  auto scatterV = [&](int buf) {
#pragma unroll
    for (int j = 0; j < 3; ++j) {
      int idx = tid + j * 256;
      if (idx < 640) {
        int key = idx / 10, c = idx % 10;
#pragma unroll
        for (int e = 0; e < 8; ++e) Vbuf[buf][(c * 8 + e) * VSTR + key] = vreg[j][e];
      }
    }
  };

  issueK(0, 0);
  loadV(0);

  for (int kb = 0; kb < N_ / 64; ++kb) {
    int cur = kb & 1;
    await_async();        // this wave's async K(kb) writes landed in LDS
    scatterV(cur);        // V(kb) regs -> LDS (other buffer than the one being computed)
    __syncthreads();      // publish K(kb)/V(kb); all waves done with block kb-1
    if (kb + 1 < N_ / 64) {
      issueK(kb + 1, cur ^ 1);  // overlap next-block staging with compute
      loadV(kb + 1);
    }
    const bf16* Kb = Kbuf[cur];
    const bf16* Vb = Vbuf[cur];

#pragma unroll
    for (int pr = 0; pr < 2; ++pr) {
      v8f S0 = {}, S1 = {};
      {
        const bf16* k0 = &Kb[((pr * 2 + 0) * 16 + col) * DQK + hi * 16];
        const bf16* k1 = &Kb[((pr * 2 + 1) * 16 + col) * DQK + hi * 16];
#pragma unroll
        for (int f = 0; f < 3; ++f) {
          S0 = wmma_bf16(qf[f], ld32(k0 + f * 32), S0);
          S1 = wmma_bf16(qf[f], ld32(k1 + f * 32), S1);
        }
      }
      float mn[8], al[8];
#pragma unroll
      for (int r = 0; r < 8; ++r) {
        S0[r] *= SM2; S1[r] *= SM2;
        float mt = redmax16(fmaxf(S0[r], S1[r]));
        mn[r] = fmaxf(mrow[r], mt);
        al[r] = exp2f(mrow[r] - mn[r]);
        mrow[r] = mn[r];
      }
#pragma unroll
      for (int r = 0; r < 8; ++r) {
        float p0 = exp2f(S0[r] - mn[r]);
        float p1 = exp2f(S1[r] - mn[r]);
        S0[r] = p0; S1[r] = p1;
        lrow[r] = lrow[r] * al[r] + redsum16(p0 + p1);
      }
#pragma unroll
      for (int v5 = 0; v5 < 5; ++v5)
#pragma unroll
        for (int r = 0; r < 8; ++r) o[v5][r] *= al[r];

      bf16* ps = &Pscr[wid * 512];
#pragma unroll
      for (int r = 0; r < 8; ++r) {
        int prow = r + 8 * hi;
        ps[prow * 32 + col]      = (bf16)S0[r];
        ps[prow * 32 + 16 + col] = (bf16)S1[r];
      }
      v16bf pf = ldA2(&ps[col * 32 + hi * 16], &ps[col * 32 + hi * 16 + 8]);
#pragma unroll
      for (int v5 = 0; v5 < 5; ++v5) {
        v16bf vf = ld32(&Vb[(v5 * 16 + col) * VSTR + pr * 32 + hi * 16]);
        o[v5] = wmma_bf16(pf, vf, o[v5]);
      }
    }
  }

  // epilogue: softmax normalize, Lorentz midpoint rescale, head-concat write
  float invl[8];
#pragma unroll
  for (int r = 0; r < 8; ++r) invl[r] = 1.0f / lrow[r];
  float fac[8];
  {
    float loc[8];
#pragma unroll
    for (int r = 0; r < 8; ++r) loc[r] = 0.0f;
#pragma unroll
    for (int v5 = 0; v5 < 5; ++v5) {
      int d = v5 * 16 + col;
#pragma unroll
      for (int r = 0; r < 8; ++r) {
        float xv = o[v5][r] * invl[r];
        o[v5][r] = xv;
        loc[r] += (d == 0) ? -xv * xv : xv * xv;
      }
    }
#pragma unroll
    for (int r = 0; r < 8; ++r) {
      float inner = redsum16(loc[r]);
      fac[r] = rsqrtf(fmaxf(-KCURV * inner, EPSF));
    }
  }
  float sp[8];
#pragma unroll
  for (int r = 0; r < 8; ++r) sp[r] = 0.0f;
#pragma unroll
  for (int v5 = 0; v5 < 5; ++v5) {
    int d = v5 * 16 + col;
#pragma unroll
    for (int r = 0; r < 8; ++r) {
      float xv = o[v5][r] * fac[r];
      o[v5][r] = xv;
      if (d >= 1 && d <= 64) sp[r] += xv * xv;
    }
  }
#pragma unroll
  for (int r = 0; r < 8; ++r) {
    float s = redsum16(sp[r]);
    int row = q0 + r + 8 * hi;
    if (col == 0) atomicAdd(&tsum[b * N_ + row], s);
    bf16* cr = &catp[((size_t)b * N_ + row) * 512 + h * DH];
#pragma unroll
    for (int v5 = 0; v5 < 5; ++v5) {
      int d = v5 * 16 + col;
      if (d >= 1 && d <= 64) cr[d - 1] = (bf16)o[v5][r];
    }
  }
}

// ---------- kernel 3: output LorentzFC GEMM ----------
// grid (N/32, 1, B), block 256 (8 waves).  rows 32, cols 512, K=544 (t' at K=543).
__global__ __launch_bounds__(256) void k_oproj(
    const bf16* __restrict__ catp, const float* __restrict__ tsum,
    const bf16* __restrict__ wot, const float* __restrict__ bo,
    float* __restrict__ out, float* __restrict__ t2sum) {
  __shared__ alignas(16) bf16 Ablk[32 * DP];
  int b = blockIdx.z;
  int r0 = blockIdx.x * 32;
  int tid = threadIdx.x;
  int wid = tid >> 5, lane = tid & 31, col = lane & 15, hi = lane >> 4;

  for (int idx = tid; idx < 32 * 64; idx += 256) {
    int row = idx >> 6, c = idx & 63;
    cp16_g2l(&Ablk[row * DP + c * 8],
             &catp[((size_t)b * N_ + r0 + row) * 512 + c * 8]);
  }
  for (int idx = tid; idx < 32 * 31; idx += 256) {  // zero K=512..542
    int row = idx / 31, j = idx % 31;
    Ablk[row * DP + 512 + j] = (bf16)0.0f;
  }
  if (tid < 32) {
    float tp = sqrtf(fmaxf(INVK + tsum[b * N_ + r0 + tid], EPSF));
    Ablk[tid * DP + 543] = (bf16)tp;
  }
  await_async();
  __syncthreads();

  int rt = wid & 1;
  int cbase = (wid >> 1) * 8;
  v8f acc[8] = {};
  for (int kk = 0; kk < 17; ++kk) {
    int kb = kk * 32;
    v16bf a = ldA2(&Ablk[(rt * 16 + col) * DP + kb + hi * 16],
                   &Ablk[(rt * 16 + col) * DP + kb + hi * 16 + 8]);
#pragma unroll
    for (int j = 0; j < 8; ++j) {
      int m = (cbase + j) * 16 + col;
      acc[j] = wmma_bf16(a, ld32(&wot[(size_t)m * DP + kb + hi * 16]), acc[j]);
    }
  }
  float sq[8];
#pragma unroll
  for (int r = 0; r < 8; ++r) sq[r] = 0.0f;
#pragma unroll
  for (int j = 0; j < 8; ++j) {
    int m = (cbase + j) * 16 + col;
    float bb = bo[m];
#pragma unroll
    for (int r = 0; r < 8; ++r) {
      int row = r0 + rt * 16 + r + 8 * hi;
      float v = acc[j][r] + bb;
      out[((size_t)b * N_ + row) * 513 + 1 + m] = v;
      sq[r] += v * v;
    }
  }
#pragma unroll
  for (int r = 0; r < 8; ++r) {
    float s = redsum16(sq[r]);
    if (col == 0) {
      int row = r0 + rt * 16 + r + 8 * hi;
      atomicAdd(&t2sum[b * N_ + row], s);
    }
  }
}

// ---------- kernel 4: final time component ----------
__global__ void k_time(const float* __restrict__ t2sum, float* __restrict__ out) {
  int i = blockIdx.x * 256 + threadIdx.x;
  if (i < B_ * N_) out[(size_t)i * 513] = sqrtf(fmaxf(INVK + t2sum[i], EPSF));
}

// ---------- host ----------
extern "C" void kernel_launch(void* const* d_in, const int* in_sizes, int n_in,
                              void* d_out, int out_size, void* d_ws, size_t ws_size,
                              hipStream_t stream) {
  (void)in_sizes; (void)n_in; (void)out_size; (void)ws_size;
  const float* x  = (const float*)d_in[0];
  const float* Wq = (const float*)d_in[1];
  const float* bq = (const float*)d_in[2];
  const float* Wk = (const float*)d_in[3];
  const float* bk = (const float*)d_in[4];
  const float* Wv = (const float*)d_in[5];
  const float* bv = (const float*)d_in[6];
  const float* Wo = (const float*)d_in[7];
  const float* bo = (const float*)d_in[8];
  float* out = (float*)d_out;

  char* ws = (char*)d_ws;
  size_t off = 0;
  auto take = [&](size_t bytes) -> char* {
    char* p = ws + off;
    off = (off + bytes + 255) & ~(size_t)255;
    return p;
  };
  bf16* xb   = (bf16*)take((size_t)B_ * N_ * DP * 2);
  bf16* wqT  = (bf16*)take((size_t)H_ * DH * DP * 2);
  bf16* wkT  = (bf16*)take((size_t)H_ * DH * DP * 2);
  bf16* wvT  = (bf16*)take((size_t)H_ * DH * DP * 2);
  bf16* wot  = (bf16*)take((size_t)512 * DP * 2);
  bf16* qt   = (bf16*)take((size_t)B_ * H_ * N_ * DQK * 2);
  bf16* ktb  = (bf16*)take((size_t)B_ * H_ * N_ * DQK * 2);
  bf16* vtb  = (bf16*)take((size_t)B_ * H_ * N_ * DV * 2);
  bf16* catp = (bf16*)take((size_t)B_ * N_ * 512 * 2);
  float* tsum  = (float*)take((size_t)B_ * N_ * 4);
  float* t2sum = (float*)take((size_t)B_ * N_ * 4);

  (void)hipMemsetAsync(tsum, 0, (size_t)B_ * N_ * 4, stream);
  (void)hipMemsetAsync(t2sum, 0, (size_t)B_ * N_ * 4, stream);

  { int tot = B_ * N_ * DP;  k_pack_x <<<(tot + 255) / 256, 256, 0, stream>>>(x, xb, tot); }
  { int tot = H_ * DH * DP;
    k_pack_w<<<(tot + 255) / 256, 256, 0, stream>>>(Wq, wqT, tot);
    k_pack_w<<<(tot + 255) / 256, 256, 0, stream>>>(Wk, wkT, tot);
    k_pack_w<<<(tot + 255) / 256, 256, 0, stream>>>(Wv, wvT, tot); }
  { int tot = 512 * DP; k_pack_wo<<<(tot + 255) / 256, 256, 0, stream>>>(Wo, wot, tot); }

  k_proj <<<dim3(N_ / 64, H_, B_), 128, 0, stream>>>(xb, wqT, wkT, wvT, bq, bk, bv, qt, ktb, vtb);
  k_attn <<<dim3(N_ / 128, H_, B_), 256, 0, stream>>>(qt, ktb, vtb, catp, tsum);
  k_oproj<<<dim3(N_ / 32, 1, B_), 256, 0, stream>>>(catp, tsum, wot, bo, out, t2sum);
  k_time <<<(B_ * N_ + 255) / 256, 256, 0, stream>>>(t2sum, out);
}
